// SpMergeAttentionLayer_55362128445581
// MI455X (gfx1250) — matrix operand order repaired and checked by reading, not compiled, not run on hardware
//
#include <hip/hip_runtime.h>
#include <hip/hip_bf16.h>

typedef __attribute__((ext_vector_type(16))) __bf16 v16bf;
typedef __attribute__((ext_vector_type(8)))  float  v8f;

#define FIN   256
#define FOUT  128
#define ALPHA 0.2f
#define EPSF  1e-8f

// ---------------------------------------------------------------------------
// Kernel 1: Wr (2,256,128) f32 row-major  ->  WrT (2,128,256) bf16
// WrT[rel][n][k] layout makes the WMMA B-tile a contiguous 32-byte run:
// lane L (n = L&15) needs K = k0 + ((L>=16)?16:0) + e, e = 0..15.
// ---------------------------------------------------------------------------
__global__ void k_prep_wrt(const float* __restrict__ Wr, __bf16* __restrict__ wrT) {
    int idx = blockIdx.x * blockDim.x + threadIdx.x;   // 0 .. 2*128*256-1
    int rel = idx >> 15;
    int rem = idx & 32767;
    int n   = rem >> 8;       // 0..127
    int k   = rem & 255;      // 0..255
    wrT[idx] = (__bf16)Wr[rel * (FIN * FOUT) + k * FOUT + n];
}

// ---------------------------------------------------------------------------
// Kernel 2: GEMM via v_wmma_f32_16x16x32_bf16.
// Block = 8 waves. Wave w: relation = w&1, M-tile = blockIdx.x*4 + (w>>1).
// Each wave: one 16-row M-tile x all 8 N-tiles for ONE relation
// -> 64 accumulator VGPRs (no spills).
// Whole WrT (128 KB bf16) is staged in LDS once per block (320 KB available
// per workgroup on CDNA5); B-tiles then come from ds_load_b128.
// A layout (16-bit A 16x32, contiguous k, same order as B): lane half hi = lane>>4,
//   element e -> K = k0 + hi*16 + e, e = 0..15
// C/D layout: VGPR r -> row = tile*16 + hi*8 + r, col = nt*16 + (lane&15).
// ---------------------------------------------------------------------------
__global__ __launch_bounds__(256) void
k_gemm_h(const float* __restrict__ node,
         const __bf16* __restrict__ wrT,
         const float* __restrict__ edge_emb,
         float* __restrict__ h_pos,
         float* __restrict__ h_neg,
         int N) {
    extern __shared__ __bf16 lwr[];          // 2*128*256 bf16 = 131072 B

    // Cooperative stage of WrT (both relations) into LDS: 8192 x 16B.
    {
        const uint4* __restrict__ s = (const uint4*)wrT;
        uint4* d = (uint4*)lwr;
        for (int i = threadIdx.x; i < (2 * FOUT * FIN) / 8; i += 256) d[i] = s[i];
    }
    __syncthreads();

    const int wave = threadIdx.x >> 5;
    const int lane = threadIdx.x & 31;
    const int rel  = wave & 1;
    const int mt   = wave >> 1;
    const int hi   = lane >> 4;              // lane half
    const int ln   = lane & 15;
    const int tile = blockIdx.x * 4 + mt;
    const int numTiles = (N + 15) >> 4;
    if (tile >= numTiles) return;            // uniform per wave, after barrier

    const int rowA  = tile * 16 + ln;
    const int rowAc = rowA < N ? rowA : (N - 1);
    const __bf16* __restrict__ lb = lwr + (size_t)rel * (FOUT * FIN);

    v8f acc[8];
    const v8f vzero = {0.f, 0.f, 0.f, 0.f, 0.f, 0.f, 0.f, 0.f};
#pragma unroll
    for (int t = 0; t < 8; ++t) acc[t] = vzero;

    for (int kc = 0; kc < 8; ++kc) {         // K = 256 in chunks of 32
        const int k0  = kc * 32;
        const int kbA = k0 + hi * 16;
        const float4* pa  = (const float4*)(node + (size_t)rowAc * FIN + kbA);
        const float4* pa2 = (const float4*)(node + (size_t)rowAc * FIN + kbA + 8);
        float4 x0 = pa[0],  x1 = pa[1];
        float4 y0 = pa2[0], y1 = pa2[1];

        v16bf av;
        av[0]  = (__bf16)x0.x; av[1]  = (__bf16)x0.y; av[2]  = (__bf16)x0.z; av[3]  = (__bf16)x0.w;
        av[4]  = (__bf16)x1.x; av[5]  = (__bf16)x1.y; av[6]  = (__bf16)x1.z; av[7]  = (__bf16)x1.w;
        av[8]  = (__bf16)y0.x; av[9]  = (__bf16)y0.y; av[10] = (__bf16)y0.z; av[11] = (__bf16)y0.w;
        av[12] = (__bf16)y1.x; av[13] = (__bf16)y1.y; av[14] = (__bf16)y1.z; av[15] = (__bf16)y1.w;

        const int kbB = k0 + hi * 16;
        v16bf bv[8];
#pragma unroll
        for (int nt = 0; nt < 8; ++nt)
            bv[nt] = *(const v16bf*)(lb + (size_t)(nt * 16 + ln) * FIN + kbB);
#pragma unroll
        for (int nt = 0; nt < 8; ++nt)
            acc[nt] = __builtin_amdgcn_wmma_f32_16x16x32_bf16(
                false, av, false, bv[nt], (short)0, acc[nt], false, false);
    }

    float* __restrict__ hout = rel ? h_neg : h_pos;
    const int rowBase = tile * 16 + hi * 8;
#pragma unroll
    for (int nt = 0; nt < 8; ++nt) {
        const int   n  = nt * 16 + ln;
        const float eg = edge_emb[rel * FOUT + n];
#pragma unroll
        for (int r = 0; r < 8; ++r) {
            const int row = rowBase + r;
            if (row < N) hout[(size_t)row * FOUT + n] = acc[nt][r] * eg;
        }
    }
}

// ---------------------------------------------------------------------------
// Kernel 3: per-row attention scalars. One wave per row; lane covers 4 cols.
// ---------------------------------------------------------------------------
__global__ void k_scores(const float* __restrict__ h_pos, const float* __restrict__ h_neg,
                         const float* __restrict__ a,
                         float* __restrict__ s_pos, float* __restrict__ d_pos,
                         float* __restrict__ s_neg, float* __restrict__ d_neg, int N) {
    const int row  = blockIdx.x * 8 + (threadIdx.x >> 5);
    const int lane = threadIdx.x & 31;
    if (row >= N) return;
    const float4 as = *(const float4*)(a + lane * 4);
    const float4 ad = *(const float4*)(a + FOUT + lane * 4);
    const float4 hp = *(const float4*)(h_pos + (size_t)row * FOUT + lane * 4);
    const float4 hn = *(const float4*)(h_neg + (size_t)row * FOUT + lane * 4);
    float sp = hp.x*as.x + hp.y*as.y + hp.z*as.z + hp.w*as.w;
    float dp = hp.x*ad.x + hp.y*ad.y + hp.z*ad.z + hp.w*ad.w;
    float sn = hn.x*as.x + hn.y*as.y + hn.z*as.z + hn.w*as.w;
    float dn = hn.x*ad.x + hn.y*ad.y + hn.z*ad.z + hn.w*ad.w;
#pragma unroll
    for (int off = 16; off > 0; off >>= 1) {
        sp += __shfl_xor(sp, off, 32);
        dp += __shfl_xor(dp, off, 32);
        sn += __shfl_xor(sn, off, 32);
        dn += __shfl_xor(dn, off, 32);
    }
    if (lane == 0) { s_pos[row] = sp; d_pos[row] = dp; s_neg[row] = sn; d_neg[row] = dn; }
}

// ---------------------------------------------------------------------------
// Kernel 4: per-edge score + scatter-add. 256-thread block = 2 edges,
// thread covers one output feature. agg lives in d_out; row_sum in workspace.
// h buffers (102 MB) + agg (51 MB) fit in the 192 MB L2 -> atomics stay on-chip.
// ---------------------------------------------------------------------------
__global__ void k_edges(const int* __restrict__ adj_pos, const int* __restrict__ adj_neg,
                        const float* __restrict__ h_pos, const float* __restrict__ h_neg,
                        const float* __restrict__ s_pos, const float* __restrict__ d_pos,
                        const float* __restrict__ s_neg, const float* __restrict__ d_neg,
                        float* __restrict__ agg, float* __restrict__ row_sum, int E) {
    const int eid = blockIdx.x * 2 + (threadIdx.x >> 7);
    if (eid >= 2 * E) return;
    const int rel = (eid >= E) ? 1 : 0;
    const int i   = rel ? (eid - E) : eid;
    const int* __restrict__ adj = rel ? adj_neg : adj_pos;
    const int src = adj[i];
    const int dst = adj[E + i];
    float z = (rel ? s_neg : s_pos)[src] + (rel ? d_neg : d_pos)[dst];
    z = (z > 0.f) ? z : (ALPHA * z);
    const float e = 1.f / (1.f + __expf(-z));
    const float* __restrict__ h = rel ? h_neg : h_pos;
    const int f = threadIdx.x & (FOUT - 1);
    atomicAdd(agg + (size_t)src * FOUT + f, e * h[(size_t)dst * FOUT + f]);
    if (f == 0) atomicAdd(row_sum + src, e);
}

// ---------------------------------------------------------------------------
// Kernel 5: in-place normalize + bias.
// ---------------------------------------------------------------------------
__global__ void k_final(float* __restrict__ out, const float* __restrict__ row_sum,
                        const float* __restrict__ bias, int total) {
    const int idx = blockIdx.x * blockDim.x + threadIdx.x;
    if (idx >= total) return;
    const int m = idx >> 7;
    const int f = idx & (FOUT - 1);
    out[idx] = out[idx] / (row_sum[m] + EPSF) + bias[f];
}

// ---------------------------------------------------------------------------
extern "C" void kernel_launch(void* const* d_in, const int* in_sizes, int n_in,
                              void* d_out, int out_size, void* d_ws, size_t ws_size,
                              hipStream_t stream) {
    const float* node     = (const float*)d_in[0];   // (N, 256)
    const float* edge_emb = (const float*)d_in[1];   // (2, 128)
    const float* Wr       = (const float*)d_in[2];   // (2, 256, 128)
    const float* a        = (const float*)d_in[3];   // (1, 256)
    const float* bias     = (const float*)d_in[4];   // (1, 128)
    const int*   adj_pos  = (const int*)d_in[5];     // (2, E)
    const int*   adj_neg  = (const int*)d_in[6];     // (2, E)

    const int N = in_sizes[0] / FIN;
    const int E = in_sizes[5] / 2;

    // Workspace layout (all offsets 32B-aligned).
    char*   ws    = (char*)d_ws;
    size_t  off   = 0;
    float*  h_pos = (float*)(ws + off); off += (size_t)N * FOUT * sizeof(float);
    float*  h_neg = (float*)(ws + off); off += (size_t)N * FOUT * sizeof(float);
    __bf16* wrT   = (__bf16*)(ws + off); off += (size_t)2 * FIN * FOUT * sizeof(__bf16);
    float*  s_pos = (float*)(ws + off); off += (size_t)N * sizeof(float);
    float*  d_pos = (float*)(ws + off); off += (size_t)N * sizeof(float);
    float*  s_neg = (float*)(ws + off); off += (size_t)N * sizeof(float);
    float*  d_neg = (float*)(ws + off); off += (size_t)N * sizeof(float);
    float*  rsum  = (float*)(ws + off); off += (size_t)N * sizeof(float);

    // Zero the atomic accumulators (graph-capturable memset nodes).
    hipMemsetAsync(d_out, 0, (size_t)N * FOUT * sizeof(float), stream);
    hipMemsetAsync(rsum,  0, (size_t)N * sizeof(float), stream);

    // 1) Wr -> bf16, transposed to WMMA B layout.
    k_prep_wrt<<<(2 * FIN * FOUT + 255) / 256, 256, 0, stream>>>(Wr, wrT);

    // 2) GEMM via WMMA (8 waves/block: 4 M-tiles x 2 relations), 128 KB LDS.
    const int numTiles = (N + 15) / 16;
    const size_t ldsBytes = (size_t)2 * FOUT * FIN * sizeof(__bf16);
    k_gemm_h<<<(numTiles + 3) / 4, 256, ldsBytes, stream>>>(node, wrT, edge_emb,
                                                            h_pos, h_neg, N);

    // 3) Row attention scalars.
    k_scores<<<(N + 7) / 8, 256, 0, stream>>>(h_pos, h_neg, a,
                                              s_pos, d_pos, s_neg, d_neg, N);

    // 4) Edge scatter (pos edges then neg edges in one grid, 2 edges/block).
    k_edges<<<E, 256, 0, stream>>>(adj_pos, adj_neg, h_pos, h_neg,
                                   s_pos, d_pos, s_neg, d_neg,
                                   (float*)d_out, rsum, E);

    // 5) Normalize + bias in place.
    const int total = N * FOUT;
    k_final<<<(total + 255) / 256, 256, 0, stream>>>((float*)d_out, rsum, bias, total);
}
